// ClusteringLayer_88940182766172
// MI455X (gfx1250) — matrix operand order repaired and checked by reading, not compiled, not run on hardware
//
#include <hip/hip_runtime.h>

typedef __attribute__((ext_vector_type(16))) __bf16 v16bf;
typedef __attribute__((ext_vector_type(8)))  float  v8f;

#define N_PTS   65536
#define K_CL    256
#define D_DIM   512
#define WAVES_PER_BLOCK 8
#define ROWS_PER_WAVE   16
#define ROWS_PER_BLOCK  (WAVES_PER_BLOCK * ROWS_PER_WAVE)   // 128
#define KSTEPS  (D_DIM / 32)                                // 16
#define NTILES  (K_CL / 16)                                 // 16

#define LDS_CB_BYTES (K_CL * D_DIM * 2)                     // 262144 B bf16 clusters
#define LDS_C2_OFF   LDS_CB_BYTES                           // 256 floats c2
#define LDS_X2_OFF   (LDS_C2_OFF + K_CL * 4)                // 8*32 floats x2 partials
#define LDS_BYTES    (LDS_X2_OFF + WAVES_PER_BLOCK * 32 * 4)  // 264192 B <= 320 KB

union BF16x16 { v16bf v; unsigned w[8]; };

// fp32 -> bf16 (RNE) pack of two floats into one dword using CDNA5 VOP3P mix ops:
// V_FMA_MIXLO_BF16 / V_FMA_MIXHI_BF16 (ISA 15.10 opcodes 62/63). fma(a,1.0,0) == a,
// converted+rounded to bf16 into the low/high half; MIXHI preserves the other half.
__device__ __forceinline__ unsigned pk_bf16(float a, float b) {
  unsigned d;
  asm("v_fma_mixlo_bf16 %0, %1, 1.0, 0" : "=v"(d) : "v"(a));
  asm("v_fma_mixhi_bf16 %0, %1, 1.0, 0" : "+v"(d) : "v"(b));
  return d;
}

__global__ __launch_bounds__(256)
void cluster_q_kernel(const float* __restrict__ X,
                      const float* __restrict__ C,
                      float* __restrict__ Q) {
  extern __shared__ char smem[];
  __bf16* cb  = (__bf16*)smem;                 // [K_CL][D_DIM] bf16, row-major
  float*  c2  = (float*)(smem + LDS_C2_OFF);   // [K_CL]
  float*  x2s = (float*)(smem + LDS_X2_OFF);   // [8][32]

  const int tid = threadIdx.x;

  // ---- Stage clusters into LDS as bf16; compute c2[k] (one cluster row per thread)
  {
    const float4* src = (const float4*)(C + tid * D_DIM);
    float s = 0.f;
    #pragma unroll 4
    for (int i = 0; i < D_DIM / 4; ++i) {
      float4 v = src[i];
      s = fmaf(v.x, v.x, s); s = fmaf(v.y, v.y, s);
      s = fmaf(v.z, v.z, s); s = fmaf(v.w, v.w, s);
      uint2 p;
      p.x = pk_bf16(v.x, v.y);
      p.y = pk_bf16(v.z, v.w);
      *(uint2*)(smem + (tid * D_DIM + i * 4) * 2) = p;
    }
    c2[tid] = s;
  }
  __syncthreads();

  const int lane = tid & 31;
  const int wave = tid >> 5;
  const int m    = lane & 15;        // A row / B column within tile
  const int half = lane >> 4;        // which 16-lane half
  const int rowbase = blockIdx.x * ROWS_PER_BLOCK + wave * ROWS_PER_WAVE;
  const float* xrow = X + (size_t)(rowbase + m) * D_DIM;

  v8f acc[NTILES];
  #pragma unroll
  for (int t = 0; t < NTILES; ++t)
    #pragma unroll
    for (int j = 0; j < 8; ++j) acc[t][j] = 0.f;

  float x2p = 0.f;   // partial ||x||^2 over this lane's K subset

  for (int kk = 0; kk < KSTEPS; ++kk) {
    // A fragment (16x32 bf16): lanes 0-15 row m, K {0..15}; lanes 16-31 row m, K {16..31}
    const int o1 = kk * 32 + half * 16;
    float4 a0 = *(const float4*)(xrow + o1);
    float4 a1 = *(const float4*)(xrow + o1 + 4);
    float4 a2 = *(const float4*)(xrow + o1 + 8);
    float4 a3 = *(const float4*)(xrow + o1 + 12);

    x2p = fmaf(a0.x, a0.x, x2p); x2p = fmaf(a0.y, a0.y, x2p);
    x2p = fmaf(a0.z, a0.z, x2p); x2p = fmaf(a0.w, a0.w, x2p);
    x2p = fmaf(a1.x, a1.x, x2p); x2p = fmaf(a1.y, a1.y, x2p);
    x2p = fmaf(a1.z, a1.z, x2p); x2p = fmaf(a1.w, a1.w, x2p);
    x2p = fmaf(a2.x, a2.x, x2p); x2p = fmaf(a2.y, a2.y, x2p);
    x2p = fmaf(a2.z, a2.z, x2p); x2p = fmaf(a2.w, a2.w, x2p);
    x2p = fmaf(a3.x, a3.x, x2p); x2p = fmaf(a3.y, a3.y, x2p);
    x2p = fmaf(a3.z, a3.z, x2p); x2p = fmaf(a3.w, a3.w, x2p);

    BF16x16 af;
    af.w[0] = pk_bf16(a0.x, a0.y); af.w[1] = pk_bf16(a0.z, a0.w);
    af.w[2] = pk_bf16(a1.x, a1.y); af.w[3] = pk_bf16(a1.z, a1.w);
    af.w[4] = pk_bf16(a2.x, a2.y); af.w[5] = pk_bf16(a2.z, a2.w);
    af.w[6] = pk_bf16(a3.x, a3.y); af.w[7] = pk_bf16(a3.z, a3.w);
    const v16bf av = af.v;

    // B fragment (32x16 bf16): lane = column n, K contiguous per half
    const int kb = kk * 32 + half * 16;
    #pragma unroll
    for (int t = 0; t < NTILES; ++t) {
      const int n = t * 16 + m;
      v16bf bv = *(const v16bf*)(cb + n * D_DIM + kb);
      acc[t] = __builtin_amdgcn_wmma_f32_16x16x32_bf16(
          false, av, false, bv, (short)0, acc[t], false, false);
    }
  }

  // ---- exchange ||x||^2 partials: row m total = lane m partial + lane m+16 partial
  x2s[wave * 32 + lane] = x2p;
  float xr[8];
  #pragma unroll
  for (int r = 0; r < 8; ++r) {
    const int mr = half * 8 + r;          // C/D layout: row M = r + 8*half
    xr[r] = x2s[wave * 32 + mr] + x2s[wave * 32 + 16 + mr];
  }

  // ---- epilogue: q = 1/(1 + max(x2 + c2 - 2*xc, 0)); row-normalize over K
  float rs[8];
  #pragma unroll
  for (int r = 0; r < 8; ++r) rs[r] = 0.f;

  #pragma unroll
  for (int t = 0; t < NTILES; ++t) {
    const float c2v = c2[t * 16 + m];
    #pragma unroll
    for (int r = 0; r < 8; ++r) {
      float dist = fmaf(-2.f, acc[t][r], xr[r] + c2v);
      dist = fmaxf(dist, 0.f);
      const float q = __builtin_amdgcn_rcpf(1.f + dist);   // v_rcp_f32, 1 ulp
      acc[t][r] = q;
      rs[r] += q;
    }
  }

  // reduce row sums across each 16-lane half (masks < 16 stay within a half)
  #pragma unroll
  for (int mask = 1; mask < 16; mask <<= 1)
    #pragma unroll
    for (int r = 0; r < 8; ++r)
      rs[r] += __shfl_xor(rs[r], mask, 32);

  #pragma unroll
  for (int r = 0; r < 8; ++r) rs[r] = __builtin_amdgcn_rcpf(rs[r]);

  #pragma unroll
  for (int t = 0; t < NTILES; ++t) {
    const int col = t * 16 + m;
    #pragma unroll
    for (int r = 0; r < 8; ++r) {
      const int row = rowbase + half * 8 + r;
      Q[(size_t)row * K_CL + col] = acc[t][r] * rs[r];
    }
  }
}

extern "C" void kernel_launch(void* const* d_in, const int* in_sizes, int n_in,
                              void* d_out, int out_size, void* d_ws, size_t ws_size,
                              hipStream_t stream) {
  (void)in_sizes; (void)n_in; (void)out_size; (void)d_ws; (void)ws_size;
  const float* X = (const float*)d_in[0];   // [N_PTS, D_DIM] fp32
  const float* C = (const float*)d_in[1];   // [K_CL,  D_DIM] fp32
  float* Q = (float*)d_out;                 // [N_PTS, K_CL]  fp32

  dim3 grid(N_PTS / ROWS_PER_BLOCK);        // 512 blocks
  dim3 block(256);                          // 8 wave32 waves
  cluster_q_kernel<<<grid, block, LDS_BYTES, stream>>>(X, C, Q);
}
